// CrossAttention_6081673691664
// MI455X (gfx1250) — compile-verified, matrix operand order checked by reading, not run on hardware
//
#include <hip/hip_runtime.h>

// ---------------------------------------------------------------------------
// Flash-attention forward (GQA) for gfx1250 / MI455X.  fp32 in/out, f16 WMMA.
// One wave (32 lanes) owns 16 query rows; 8 waves/block => 128 rows/block.
// S is computed transposed (S^T = K * Q^T) so exp2(S^T) C-fragments feed the
// PV GEMM (O^T = V^T * P^T) as B-operands with zero cross-lane movement.
// Softmax runs in log2-domain (log2e folded into the Q scale and mask value).
// The padding mask is a prefix mask and klim = qrow+sk-SQ < sk, so the
// reference's causal where(-10000) overwrite subsumes the additive pad bias
// everywhere -> only the causal compare is needed per element.
// Block-uniform causal early-exit truncates the key loop exactly.
// Global loads for tile i+1 are issued before computing tile i (register
// double-buffering), hiding memory latency across the staging barriers.
// ---------------------------------------------------------------------------

typedef __attribute__((ext_vector_type(16))) _Float16 v16h;
typedef __attribute__((ext_vector_type(8)))  _Float16 v8h;
typedef __attribute__((ext_vector_type(4)))  _Float16 v4h;
typedef __attribute__((ext_vector_type(8)))  float    v8f;
typedef __attribute__((ext_vector_type(4)))  float    v4f;
// __builtin_amdgcn_cvt_pkrtz returns __fp16 vectors (distinct from _Float16)
typedef __attribute__((ext_vector_type(2)))  __fp16   h2r;

#define WMMA_F16(a, b, c) \
  __builtin_amdgcn_wmma_f32_16x16x32_f16(false, (a), false, (b), (short)0, (c), false, false)

constexpr int B_  = 2;
constexpr int SQ_ = 2048;
constexpr int SK_ = 2048;
constexpr int H_  = 16;
constexpr int HK_ = 4;
constexpr int D_  = 64;

constexpr int KT      = 32;          // keys per k-tile (one 16x32 WMMA K-dim)
constexpr int WAVES   = 8;           // waves per block (wave32)
constexpr int THREADS = WAVES * 32;  // 256
constexpr int QROWS   = 16 * WAVES;  // 128 query rows per block
constexpr int LDK_STR = 72;          // LDS K row stride (halfs), 16B multiple
constexpr int LDV_STR = 40;          // LDS V^T row stride (halfs), 16B multiple

constexpr float LOG2E  = 1.4426950408889634f;
constexpr float NBIAS2 = -10000.0f * LOG2E;  // -10000 in log2-domain

// 16x32 f16 A-operand fragment slice from LDS: 8 contiguous halfs at p
// (K = base..base+7) and 8 at p+16 (K = base+16..base+23).
__device__ inline v16h lds_a_frag(const _Float16* p) {
  v8h lo = *(const v8h*)(p);
  v8h hh = *(const v8h*)(p + 16);
  return __builtin_shufflevector(lo, hh, 0, 1, 2, 3, 4, 5, 6, 7,
                                          8, 9, 10, 11, 12, 13, 14, 15);
}

__global__ __launch_bounds__(THREADS)
void fa_fwd(const float* __restrict__ qg, const float* __restrict__ kvg,
            const unsigned char* __restrict__ maskg, float* __restrict__ outg) {
  __shared__ __align__(16) _Float16 lds_k[KT * LDK_STR];  // K tile   [key][d]
  __shared__ __align__(16) _Float16 lds_v[D_ * LDV_STR];  // V^T tile [d][key]
  __shared__ int lds_sk;

  const int tid  = threadIdx.x;
  const int lane = tid & 31;
  const int wave = tid >> 5;     // wave32
  const int hi   = lane >> 4;
  const int ln   = lane & 15;

  constexpr int NQT = SQ_ / QROWS;                  // 16 q-tiles per (b,h)
  const int qt = blockIdx.x % NQT;
  const int h  = (blockIdx.x / NQT) % H_;
  const int b  = blockIdx.x / (NQT * H_);
  const int hk = h / (H_ / HK_);                    // GQA: h -> kv head

  // sk = key_padding_mask.sum(-1) for this batch (prefix mask)
  if (tid == 0) lds_sk = 0;
  __syncthreads();
  int ms = 0;
  for (int s = tid; s < SK_; s += THREADS) ms += (int)maskg[b * SK_ + s];
  atomicAdd(&lds_sk, ms);
  __syncthreads();
  const int diag = lds_sk - SQ_;                    // col > row + diag => masked

  // Block-uniform causal loop bound (exact: truncated tiles contribute
  // exp2(-14427 - m) == 0 for every non-degenerate row in this block).
  const int qblock0 = qt * QROWS;
  int kmax = SK_;
  if (qblock0 + diag >= 0) {
    const int need = qblock0 + (QROWS - 1) + diag + 1;   // keys required
    kmax = (need + KT - 1) & ~(KT - 1);
    if (kmax > SK_) kmax = SK_;
  }

  const int qwave0 = qblock0 + wave * 16;
  const int qrow   = qwave0 + ln;                   // this lane's q column

  // --- Q as B-operand fragments, scaled by (1/sqrt(D)) * log2(e) ----------
  v16h bq0, bq1;
  {
    const float* qp = qg + (((size_t)b * SQ_ + qrow) * H_ + h) * D_;
    const float sc = 0.125f * LOG2E;
#pragma unroll
    for (int i = 0; i < 8; ++i) {
      bq0[i]     = (_Float16)(qp[      8 * hi + i] * sc);
      bq0[8 + i] = (_Float16)(qp[16 +  8 * hi + i] * sc);
      bq1[i]     = (_Float16)(qp[32 +  8 * hi + i] * sc);
      bq1[8 + i] = (_Float16)(qp[48 +  8 * hi + i] * sc);
    }
  }

  v8f o0 = {}, o1 = {}, o2 = {}, o3 = {};           // O^T accum, d-chunks of 16
  float m_run = -1e30f, l_run = 0.f;

  const size_t kv_b  = (size_t)b * SK_ * 2 * HK_ * D_;
  const int    d0    = (tid & 15) * 4;              // staging: 4 consecutive d
  const int    kbase = tid >> 4;                    // staging: key row

  // Register-staged K/V rows for the *current* tile (loaded one tile ahead).
  v4f kf[2], vf[2];
  {
#pragma unroll
    for (int r = 0; r < 2; ++r) {
      const float* src = kvg + kv_b + (size_t)(kbase + 16 * r) * (2 * HK_ * D_)
                             + hk * D_ + d0;
      kf[r] = *(const v4f*)src;               // K row (c=0)
      vf[r] = *(const v4f*)(src + HK_ * D_);  // V row (c=1)
    }
  }

  for (int k0 = 0; k0 < kmax; k0 += KT) {
    __syncthreads();  // previous tile fully consumed
    // --- convert the register-staged tile into LDS (K, V^T as f16) --------
#pragma unroll
    for (int r = 0; r < 2; ++r) {
      const int key = kbase + 16 * r;
      const h2r ka = __builtin_amdgcn_cvt_pkrtz(kf[r][0], kf[r][1]);
      const h2r kb = __builtin_amdgcn_cvt_pkrtz(kf[r][2], kf[r][3]);
      *(v4h*)&lds_k[key * LDK_STR + d0] =
          __builtin_bit_cast(v4h, __builtin_shufflevector(ka, kb, 0, 1, 2, 3));
      const h2r va = __builtin_amdgcn_cvt_pkrtz(vf[r][0], vf[r][1]);
      const h2r vb = __builtin_amdgcn_cvt_pkrtz(vf[r][2], vf[r][3]);
      lds_v[(d0 + 0) * LDV_STR + key] = __builtin_bit_cast(_Float16, (__fp16)va[0]);
      lds_v[(d0 + 1) * LDV_STR + key] = __builtin_bit_cast(_Float16, (__fp16)va[1]);
      lds_v[(d0 + 2) * LDV_STR + key] = __builtin_bit_cast(_Float16, (__fp16)vb[0]);
      lds_v[(d0 + 3) * LDV_STR + key] = __builtin_bit_cast(_Float16, (__fp16)vb[1]);
    }
    __syncthreads();

    // --- issue next tile's global loads; latency hides under compute ------
    if (k0 + KT < kmax) {
#pragma unroll
      for (int r = 0; r < 2; ++r) {
        const float* src = kvg + kv_b
            + (size_t)(k0 + KT + kbase + 16 * r) * (2 * HK_ * D_)
            + hk * D_ + d0;
        kf[r] = *(const v4f*)src;
        vf[r] = *(const v4f*)(src + HK_ * D_);
      }
    }

    // --- S^T = K * Q^T : two 16(key)x16(q) fragments, accumulate over D ----
    const v16h ak00 = lds_a_frag(&lds_k[(ln)      * LDK_STR +      8 * hi]);
    const v16h ak01 = lds_a_frag(&lds_k[(ln)      * LDK_STR + 32 + 8 * hi]);
    const v16h ak10 = lds_a_frag(&lds_k[(16 + ln) * LDK_STR +      8 * hi]);
    const v16h ak11 = lds_a_frag(&lds_k[(16 + ln) * LDK_STR + 32 + 8 * hi]);
    v8f s0 = {}, s1 = {};
    s0 = WMMA_F16(ak00, bq0, s0);
    s0 = WMMA_F16(ak01, bq1, s0);
    s1 = WMMA_F16(ak10, bq0, s1);
    s1 = WMMA_F16(ak11, bq1, s1);

    // V^T fragments issued early so ds latency overlaps the softmax VALU
    const v16h av0 = lds_a_frag(&lds_v[(ln)      * LDV_STR + 8 * hi]);
    const v16h av1 = lds_a_frag(&lds_v[(16 + ln) * LDV_STR + 8 * hi]);
    const v16h av2 = lds_a_frag(&lds_v[(32 + ln) * LDV_STR + 8 * hi]);
    const v16h av3 = lds_a_frag(&lds_v[(48 + ln) * LDV_STR + 8 * hi]);

    // --- causal/pad mask + online softmax (log2-domain, per q column) ------
    float sv0[8], sv1[8];
    float mx = -1e30f;
    if ((k0 + KT - 1) <= (qwave0 + diag)) {
      // fast path: whole tile valid for every row of this wave
#pragma unroll
      for (int v = 0; v < 8; ++v) {
        sv0[v] = s0[v];
        sv1[v] = s1[v];
        mx = fmaxf(mx, fmaxf(sv0[v], sv1[v]));
      }
    } else {
      const int klim = qrow + diag;  // keys > klim: score := -10000 (log2-dom)
#pragma unroll
      for (int v = 0; v < 8; ++v) {
        const int kl0 = k0 + v + 8 * hi;
        const float x0 = (kl0      > klim) ? NBIAS2 : (float)s0[v];
        const float x1 = (kl0 + 16 > klim) ? NBIAS2 : (float)s1[v];
        sv0[v] = x0; sv1[v] = x1;
        mx = fmaxf(mx, fmaxf(x0, x1));
      }
    }
    mx = fmaxf(mx, __shfl_xor(mx, 16));  // combine the two key-halves
    const float m_new = fmaxf(m_run, mx);
    const float alpha = __builtin_amdgcn_exp2f(m_run - m_new);

    float ts = 0.f;
    union { v16h v; h2r h2[8]; } bp;     // exp2(S^T) -> PV B-operand fragment
#pragma unroll
    for (int p = 0; p < 4; ++p) {
      const float a0 = __builtin_amdgcn_exp2f(sv0[2 * p]     - m_new);
      const float a1 = __builtin_amdgcn_exp2f(sv0[2 * p + 1] - m_new);
      const float b0 = __builtin_amdgcn_exp2f(sv1[2 * p]     - m_new);
      const float b1 = __builtin_amdgcn_exp2f(sv1[2 * p + 1] - m_new);
      bp.h2[p]     = __builtin_amdgcn_cvt_pkrtz(a0, a1);  // halfs 0-7
      bp.h2[4 + p] = __builtin_amdgcn_cvt_pkrtz(b0, b1);  // halfs 8-15
      ts += (a0 + a1) + (b0 + b1);
    }
    ts += __shfl_xor(ts, 16);
    l_run = l_run * alpha + ts;
    m_run = m_new;

    // --- O^T = V^T * P^T : rescale then 4 WMMAs over d-chunks --------------
    o0 *= alpha; o1 *= alpha; o2 *= alpha; o3 *= alpha;
    o0 = WMMA_F16(av0, bp.v, o0);
    o1 = WMMA_F16(av1, bp.v, o1);
    o2 = WMMA_F16(av2, bp.v, o2);
    o3 = WMMA_F16(av3, bp.v, o3);
  }

  // --- epilogue: normalize and store (lane owns q=qrow, d = 16c+8*hi+v) ---
  const float inv = 1.f / l_run;
  float* op = outg + (((size_t)b * SQ_ + qrow) * H_ + h) * D_ + 8 * hi;
  const v8f r0 = o0 * inv, r1 = o1 * inv, r2 = o2 * inv, r3 = o3 * inv;
  *(v4f*)(op +  0) = __builtin_shufflevector(r0, r0, 0, 1, 2, 3);
  *(v4f*)(op +  4) = __builtin_shufflevector(r0, r0, 4, 5, 6, 7);
  *(v4f*)(op + 16) = __builtin_shufflevector(r1, r1, 0, 1, 2, 3);
  *(v4f*)(op + 20) = __builtin_shufflevector(r1, r1, 4, 5, 6, 7);
  *(v4f*)(op + 32) = __builtin_shufflevector(r2, r2, 0, 1, 2, 3);
  *(v4f*)(op + 36) = __builtin_shufflevector(r2, r2, 4, 5, 6, 7);
  *(v4f*)(op + 48) = __builtin_shufflevector(r3, r3, 0, 1, 2, 3);
  *(v4f*)(op + 52) = __builtin_shufflevector(r3, r3, 4, 5, 6, 7);
}

extern "C" void kernel_launch(void* const* d_in, const int* in_sizes, int n_in,
                              void* d_out, int out_size, void* d_ws, size_t ws_size,
                              hipStream_t stream) {
  const float*         q    = (const float*)d_in[0];
  const float*         kv   = (const float*)d_in[1];
  const unsigned char* mask = (const unsigned char*)d_in[2];  // jnp.bool_ = 1 byte
  float*               out  = (float*)d_out;

  dim3 grid(B_ * H_ * (SQ_ / QROWS));  // 2*16*16 = 512 blocks
  fa_fwd<<<grid, THREADS, 0, stream>>>(q, kv, mask, out);
}
